// OverlapAggregator_1743756722879
// MI455X (gfx1250) — matrix operand order repaired and checked by reading, not compiled, not run on hardware
//
#include <hip/hip_runtime.h>
#include <hip/hip_bf16.h>
#include <math.h>
#include <stdint.h>

typedef unsigned short u16;
typedef __attribute__((ext_vector_type(16))) __bf16 v16bf;
typedef __attribute__((ext_vector_type(8)))  __bf16 v8bf;
typedef __attribute__((ext_vector_type(8)))  float  v8f;
typedef __attribute__((ext_vector_type(4)))  unsigned int tv4u;
typedef __attribute__((ext_vector_type(4)))  int tv4i;
typedef __attribute__((ext_vector_type(8)))  int tv8i;

#define B_    2
#define T_    1024
#define D_    1024
#define H_    16
#define DH_   64
#define KV_   2048
#define KP_   2304   /* gate-in K padded 2051 -> 2304 (9 x 256) */
#define GIN_  2051
#define EPS_  1e-5f
#define CHUNK_ 256   /* K-chunk staged to LDS by the TDM */
#define LROW_  264   /* LDS row stride in u16: 256 data + 8 pad (TDM pad: 4dw/128dw) */
#define LDSB_  (32 * LROW_ * 2)   /* 16896 bytes */

#if defined(__has_include)
#if __has_include(<hip/amd_detail/amd_gfx1250_TDM.h>)
#define TDM_6ARG 1
#endif
#endif

// ---------- helpers ----------

__device__ inline u16 f2bf(float f){
  unsigned u = __float_as_uint(f);
  u += 0x7FFFu + ((u >> 16) & 1u);          // round-to-nearest-even
  return (u16)(u >> 16);
}

__device__ inline v16bf ld16g(const u16* lo, const u16* hi){
  v8bf a = *reinterpret_cast<const v8bf*>(lo);
  v8bf b = *reinterpret_cast<const v8bf*>(hi);
  v16bf r;
#pragma unroll
  for (int i = 0; i < 8; ++i){ r[i] = a[i]; r[i + 8] = b[i]; }
  return r;
}

__device__ inline v8f wmma_bf16(v16bf a, v16bf b, v8f c){
  return __builtin_amdgcn_wmma_f32_16x16x32_bf16(
      /*neg_a=*/false, a, /*neg_b=*/false, b,
      /*c_mod=*/(short)0, c, /*reuse_a=*/false, /*reuse_b=*/false);
}

__device__ inline float gelu_exact(float x){
  return 0.5f * x * (1.f + erff(x * 0.70710678118654752f));
}

// TDM: load a rows x cols (bf16) tile of W (row stride = strideK elements) into LDS
// at byte offset lds_off, padding 4 dwords after every 128-dword (= one 256-col) row.
__device__ inline void tdm_load_w(const u16* gptr, unsigned lds_off,
                                  unsigned rows, unsigned cols, unsigned strideK){
  unsigned long long ga = (unsigned long long)(uintptr_t)gptr;
  tv4u g0;
  g0[0] = 1u;                                          // count=1, user descriptor
  g0[1] = lds_off;                                     // lds_addr (bytes)
  g0[2] = (unsigned)(ga & 0xFFFFFFFFull);              // global_addr[31:0]
  g0[3] = (unsigned)((ga >> 32) & 0x01FFFFFFull) | 0x80000000u; // addr[56:32] | type=2
  tv8i g1;
  g1[0] = (int)0x07910000u;   // data_size=2B | pad_enable | pad_interval=128dw | pad_amount=4dw
  g1[1] = (int)((strideK & 0xFFFFu) << 16);            // tensor_dim0[15:0]
  g1[2] = (int)((strideK >> 16) | ((rows & 0xFFFFu) << 16)); // dim0[31:16] | tensor_dim1[15:0]
  g1[3] = (int)((cols & 0xFFFFu) << 16);               // tensor_dim1[31:16]=0 | tile_dim0
  g1[4] = (int)(rows & 0xFFFFu);                       // tile_dim1 | tile_dim2=0
  g1[5] = (int)strideK;                                // tensor_dim0_stride[31:0]
  g1[6] = 0;                                           // stride0[47:32] | stride1[15:0]
  g1[7] = 0;
  tv4i gz = {0, 0, 0, 0};
#ifdef TDM_6ARG
  tv8i gz8 = {0, 0, 0, 0, 0, 0, 0, 0};
  __builtin_amdgcn_tensor_load_to_lds(g0, g1, gz, gz, gz8, 0);
#else
  __builtin_amdgcn_tensor_load_to_lds(g0, g1, gz, gz, 0);
#endif
}

// block-wide sum of two values; assumes blockDim.x == 256 (8 waves, wave32)
__device__ inline void block_sum2(float& a, float& b){
#pragma unroll
  for (int m = 16; m >= 1; m >>= 1){
    a += __shfl_xor(a, m, 32);
    b += __shfl_xor(b, m, 32);
  }
  __shared__ float t[16];
  int wid = threadIdx.x >> 5;
  if ((threadIdx.x & 31) == 0){ t[wid*2] = a; t[wid*2+1] = b; }
  __syncthreads();
  float ra = 0.f, rb = 0.f;
#pragma unroll
  for (int i = 0; i < 8; ++i){ ra += t[i*2]; rb += t[i*2+1]; }
  a = ra; b = rb;
  __syncthreads();
}

// ---------- elementwise / reduction kernels ----------

__global__ void ln_bf16_kernel(const float* __restrict__ x, const float* __restrict__ w,
                               const float* __restrict__ bb, u16* __restrict__ y){
  int row = blockIdx.x;
  const float* xr = x + (size_t)row * D_;
  float s = 0.f, s2 = 0.f;
  for (int i = threadIdx.x; i < D_; i += 256){ float v = xr[i]; s += v; s2 += v*v; }
  block_sum2(s, s2);
  float mean = s * (1.f / D_);
  float inv  = rsqrtf(s2 * (1.f / D_) - mean*mean + EPS_);
  u16* yr = y + (size_t)row * D_;
  for (int i = threadIdx.x; i < D_; i += 256)
    yr[i] = f2bf((xr[i] - mean) * inv * w[i] + bb[i]);
}

__global__ void cvt_bf16_kernel(const float* __restrict__ src, u16* __restrict__ dst, int n){
  int i = blockIdx.x * 256 + threadIdx.x;
  if (i < n) dst[i] = f2bf(src[i]);
}

__global__ void pad_w1_kernel(const float* __restrict__ w1, u16* __restrict__ dst){
  int idx = blockIdx.x * 256 + threadIdx.x;
  if (idx >= D_ * KP_) return;
  int row = idx / KP_, c = idx - row * KP_;
  dst[idx] = (c < GIN_) ? f2bf(w1[(size_t)row * GIN_ + c]) : (u16)0;
}

__global__ void build_gate_in_kernel(const float* __restrict__ target,
                                     const float* __restrict__ attn,
                                     const float* __restrict__ feats,
                                     u16* __restrict__ gin){
  int idx = blockIdx.x * 256 + threadIdx.x;
  if (idx >= B_ * T_ * KP_) return;
  int row = idx / KP_, c = idx - row * KP_;
  float v;
  if      (c < D_)   v = target[(size_t)row * D_ + c];
  else if (c < 2*D_) v = attn[(size_t)row * D_ + (c - D_)];
  else if (c < GIN_) v = feats[row * 3 + (c - 2*D_)];
  else               v = 0.f;
  gin[idx] = f2bf(v);
}

__global__ void gate_kernel(const float* __restrict__ h, const float* __restrict__ w2,
                            const float* __restrict__ b2, float* __restrict__ gate){
  int row = blockIdx.x;
  const float* hr = h + (size_t)row * D_;
  float s = 0.f, dummy = 0.f;
  for (int i = threadIdx.x; i < D_; i += 256) s += hr[i] * w2[i];
  block_sum2(s, dummy);
  if (threadIdx.x == 0) gate[row] = 1.f / (1.f + __expf(-(s + b2[0])));
}

__global__ void final_kernel(const float* __restrict__ target, const float* __restrict__ attn,
                             const float* __restrict__ gate, const float* __restrict__ w,
                             const float* __restrict__ bb, float* __restrict__ out,
                             float* __restrict__ out_gate){
  int row = blockIdx.x;
  float g = gate[row];
  const float* tr = target + (size_t)row * D_;
  const float* ar = attn   + (size_t)row * D_;
  float s = 0.f, s2 = 0.f;
  for (int i = threadIdx.x; i < D_; i += 256){
    float y = (2.f - g) * tr[i] + g * ar[i];   // target + (g*attn + (1-g)*target)
    s += y; s2 += y*y;
  }
  block_sum2(s, s2);
  float mean = s * (1.f / D_);
  float inv  = rsqrtf(s2 * (1.f / D_) - mean*mean + EPS_);
  float* orow = out + (size_t)row * D_;
  for (int i = threadIdx.x; i < D_; i += 256){
    float y = (2.f - g) * tr[i] + g * ar[i];
    orow[i] = (y - mean) * inv * w[i] + bb[i];
  }
  if (threadIdx.x == 0) out_gate[row] = g;
}

// ---------- WMMA GEMM:  C[M,N] = A[M,K] (bf16) @ W[N,K]^T (bf16) + bias ----------
// Block: 8 waves, each wave owns a 64x32 macro-tile (4 M-subtiles x 2 N-subtiles).
// Block covers 512 rows x 32 cols; the shared 32 x 256 weight stripe is staged into
// LDS by the Tensor Data Mover (one wave issues, TENSORcnt wait, workgroup barrier).
// mode 0: f32 out   mode 1: bf16 out   mode 2: bf16 transposed per-batch (V -> Vt[b][n][s])
// mode 3: f32 out with exact GELU
__global__ void gemm_bf16_kernel(const u16* __restrict__ A, const u16* __restrict__ W,
                                 const float* __restrict__ bias, void* __restrict__ out,
                                 int M, int N, int K, int mode, int kvlen){
  extern __shared__ u16 lds_w[];
  int wv = threadIdx.x >> 5, lane = threadIdx.x & 31;
  int hf = lane >> 4, l16 = lane & 15;

  int nblk = N >> 5;
  int msuper = blockIdx.x / nblk;
  int n0 = (blockIdx.x - msuper * nblk) << 5;
  int m_wave = msuper * 512 + wv * 64;

  const u16* arow[4];
#pragma unroll
  for (int mi = 0; mi < 4; ++mi)
    arow[mi] = A + (size_t)(m_wave + mi*16 + l16) * K;

  v8f zero = {};
  v8f acc[4][2];
#pragma unroll
  for (int mi = 0; mi < 4; ++mi){ acc[mi][0] = zero; acc[mi][1] = zero; }

  for (int kc = 0; kc < K; kc += CHUNK_){
    if (wv == 0){
      tdm_load_w(W + (size_t)n0 * K + kc, 0u, 32u, (unsigned)CHUNK_, (unsigned)K);
      __builtin_amdgcn_s_wait_tensorcnt((short)0);
    }
    __syncthreads();
#pragma unroll 2
    for (int k0 = 0; k0 < CHUNK_; k0 += 32){
      v16bf bt[2];
#pragma unroll
      for (int nl = 0; nl < 2; ++nl){
        const u16* wr = lds_w + (size_t)(nl*16 + l16) * LROW_ + k0 + hf*16;
        bt[nl] = ld16g(wr, wr + 8);
      }
#pragma unroll
      for (int mi = 0; mi < 4; ++mi){
        v16bf at = ld16g(arow[mi] + kc + k0 + hf*16, arow[mi] + kc + k0 + hf*16 + 8);
        acc[mi][0] = wmma_bf16(at, bt[0], acc[mi][0]);
        acc[mi][1] = wmma_bf16(at, bt[1], acc[mi][1]);
      }
    }
    __syncthreads();
  }

#pragma unroll
  for (int nl = 0; nl < 2; ++nl){
    int col = n0 + nl*16 + l16;
    float bv = bias[col];
#pragma unroll
    for (int mi = 0; mi < 4; ++mi){
#pragma unroll
      for (int r = 0; r < 8; ++r){
        int row = m_wave + mi*16 + r + hf*8;            // C layout: row = vgpr + 8*half
        float v = acc[mi][nl][r] + bv;
        if (mode == 3) v = gelu_exact(v);
        if (mode == 0 || mode == 3){
          ((float*)out)[(size_t)row * N + col] = v;
        } else if (mode == 1){
          ((u16*)out)[(size_t)row * N + col] = f2bf(v);
        } else {                                        // transposed store for V
          int bb = row / kvlen, s = row - bb * kvlen;
          ((u16*)out)[((size_t)bb * N + col) * kvlen + s] = f2bf(v);
        }
      }
    }
  }
}

// ---------- flash attention: one wave per (b, h, 16-query-row tile) ----------
__global__ void attn_kernel(const u16* __restrict__ Q, const u16* __restrict__ Kb,
                            const u16* __restrict__ Vt, const int* __restrict__ mask,
                            u16* __restrict__ ctx){
  __shared__ u16 pt[4][16 * 32];                      // per-wave P staging (C->A relayout)
  int wv = threadIdx.x >> 5, lane = threadIdx.x & 31;
  int gw = blockIdx.x * 4 + wv;
  int t0 = (gw & 63) * 16;
  int h  = (gw >> 6) & 15;
  int b  = gw >> 10;
  int hf = lane >> 4, l16 = lane & 15;

  const u16* qrow = Q + ((size_t)(b*T_ + t0 + l16)) * D_ + h*DH_;
  v16bf aq0 = ld16g(qrow + hf*16,      qrow + hf*16 + 8);
  v16bf aq1 = ld16g(qrow + 32 + hf*16, qrow + 32 + hf*16 + 8);

  v8f acc[4] = {{}, {}, {}, {}};
  float mrun[8], lrun[8];
#pragma unroll
  for (int r = 0; r < 8; ++r){ mrun[r] = -1e30f; lrun[r] = 0.f; }

  for (int s0 = 0; s0 < KV_; s0 += 32){
    v8f sc[2];
#pragma unroll
    for (int sub = 0; sub < 2; ++sub){
      const u16* krow = Kb + ((size_t)(b*KV_ + s0 + sub*16 + l16)) * D_ + h*DH_;
      v16bf bk0 = ld16g(krow + hf*16,      krow + hf*16 + 8);
      v16bf bk1 = ld16g(krow + 32 + hf*16, krow + 32 + hf*16 + 8);
      v8f s = {};
      s = wmma_bf16(aq0, bk0, s);
      s = wmma_bf16(aq1, bk1, s);
      int sg = s0 + sub*16 + l16;
      bool valid = mask[b*KV_ + sg] != 0;
#pragma unroll
      for (int r = 0; r < 8; ++r) s[r] = valid ? s[r] * 0.125f : -1e30f;
      sc[sub] = s;
    }
    // online softmax: rows live across 16 lanes of one half -> xor masks 1,2,4,8 only
#pragma unroll
    for (int r = 0; r < 8; ++r){
      float v = fmaxf(sc[0][r], sc[1][r]);
#pragma unroll
      for (int m = 1; m <= 8; m <<= 1) v = fmaxf(v, __shfl_xor(v, m, 32));
      float mnew = fmaxf(mrun[r], v);
      float corr = __expf(mrun[r] - mnew);
      mrun[r] = mnew;
      float p0 = __expf(sc[0][r] - mnew);
      float p1 = __expf(sc[1][r] - mnew);
      int rowl = r + hf*8;
      pt[wv][rowl*32 + l16]      = f2bf(p0);
      pt[wv][rowl*32 + 16 + l16] = f2bf(p1);
      float rs = p0 + p1;
#pragma unroll
      for (int m = 1; m <= 8; m <<= 1) rs += __shfl_xor(rs, m, 32);
      lrun[r] = lrun[r] * corr + rs;
#pragma unroll
      for (int c = 0; c < 4; ++c) acc[c][r] *= corr;
    }
    // reload P as A-matrix (LDS is in-order within a wave: no barrier needed)
    const u16* pr = &pt[wv][l16 * 32];
    v16bf pa = ld16g(pr + hf*16, pr + hf*16 + 8);
#pragma unroll
    for (int c = 0; c < 4; ++c){
      const u16* vb = Vt + ((size_t)b*D_ + h*DH_ + c*16 + l16) * KV_ + s0 + hf*16;
      acc[c] = wmma_bf16(pa, ld16g(vb, vb + 8), acc[c]);
    }
  }
#pragma unroll
  for (int c = 0; c < 4; ++c)
#pragma unroll
    for (int r = 0; r < 8; ++r){
      int row = t0 + r + hf*8;
      float v = acc[c][r] / lrun[r];
      ctx[((size_t)(b*T_ + row)) * D_ + h*DH_ + c*16 + l16] = f2bf(v);
    }
}

// ---------- host ----------

extern "C" void kernel_launch(void* const* d_in, const int* in_sizes, int n_in,
                              void* d_out, int out_size, void* d_ws, size_t ws_size,
                              hipStream_t stream){
  (void)in_sizes; (void)n_in; (void)out_size; (void)ws_size;
  const float* target   = (const float*)d_in[0];
  const float* support  = (const float*)d_in[1];
  const float* feats    = (const float*)d_in[2];
  const int*   maskp    = (const int*)d_in[3];
  const float* ln_q_w   = (const float*)d_in[4];
  const float* ln_q_b   = (const float*)d_in[5];
  const float* ln_kv_w  = (const float*)d_in[6];
  const float* ln_kv_b  = (const float*)d_in[7];
  const float* out_ln_w = (const float*)d_in[8];
  const float* out_ln_b = (const float*)d_in[9];
  const float* in_w     = (const float*)d_in[10];
  const float* in_b     = (const float*)d_in[11];
  const float* out_w    = (const float*)d_in[12];
  const float* out_b    = (const float*)d_in[13];
  const float* g_w1     = (const float*)d_in[14];
  const float* g_b1     = (const float*)d_in[15];
  const float* g_w2     = (const float*)d_in[16];
  const float* g_b2     = (const float*)d_in[17];

  char* ws = (char*)d_ws;
  size_t off = 0;
  auto take = [&](size_t bytes) -> void* {
    off = (off + 255) & ~(size_t)255;
    void* p = ws + off; off += bytes; return p;
  };
  const int Mq  = B_ * T_;    // 2048 query rows
  const int Mkv = B_ * KV_;   // 4096 kv rows

  u16*   q_ln   = (u16*)  take((size_t)Mq  * D_  * 2);
  u16*   kv_ln  = (u16*)  take((size_t)Mkv * D_  * 2);
  u16*   w_in   = (u16*)  take((size_t)3 * D_ * D_ * 2);
  u16*   w_outb = (u16*)  take((size_t)D_ * D_ * 2);
  u16*   w1p    = (u16*)  take((size_t)D_ * KP_ * 2);
  u16*   Qb     = (u16*)  take((size_t)Mq  * D_ * 2);
  u16*   Kb     = (u16*)  take((size_t)Mkv * D_ * 2);
  u16*   Vt     = (u16*)  take((size_t)B_ * D_ * KV_ * 2);
  u16*   ctx    = (u16*)  take((size_t)Mq  * D_ * 2);
  float* attn_f = (float*)take((size_t)Mq  * D_ * 4);
  u16*   gin    = (u16*)  take((size_t)Mq  * KP_ * 2);
  float* hbuf   = (float*)take((size_t)Mq  * D_ * 4);
  float* gatev  = (float*)take((size_t)Mq * 4);

  // layernorms -> bf16 activations; weights -> bf16
  ln_bf16_kernel<<<Mq, 256, 0, stream>>>(target, ln_q_w, ln_q_b, q_ln);
  ln_bf16_kernel<<<Mkv, 256, 0, stream>>>(support, ln_kv_w, ln_kv_b, kv_ln);
  cvt_bf16_kernel<<<(3*D_*D_ + 255)/256, 256, 0, stream>>>(in_w, w_in, 3*D_*D_);
  cvt_bf16_kernel<<<(D_*D_ + 255)/256, 256, 0, stream>>>(out_w, w_outb, D_*D_);
  pad_w1_kernel<<<(D_*KP_ + 255)/256, 256, 0, stream>>>(g_w1, w1p);

  // packed QKV projections (WMMA + TDM-staged weights)
  gemm_bf16_kernel<<<(Mq/512)*(D_/32), 256, LDSB_, stream>>>(
      q_ln, w_in, in_b, Qb, Mq, D_, D_, 1, 0);
  gemm_bf16_kernel<<<(Mkv/512)*(D_/32), 256, LDSB_, stream>>>(
      kv_ln, w_in + (size_t)D_*D_, in_b + D_, Kb, Mkv, D_, D_, 1, 0);
  gemm_bf16_kernel<<<(Mkv/512)*(D_/32), 256, LDSB_, stream>>>(
      kv_ln, w_in + (size_t)2*D_*D_, in_b + 2*D_, Vt, Mkv, D_, D_, 2, KV_);

  // flash attention (WMMA QK^T + PV)
  attn_kernel<<<(B_*H_*(T_/16))/4, 128, 0, stream>>>(Qb, Kb, Vt, maskp, ctx);

  // out projection (WMMA, f32 out)
  gemm_bf16_kernel<<<(Mq/512)*(D_/32), 256, LDSB_, stream>>>(
      ctx, w_outb, out_b, attn_f, Mq, D_, D_, 0, 0);

  // gate MLP: concat -> gelu GEMM (WMMA) -> sigmoid dot
  build_gate_in_kernel<<<(Mq*KP_ + 255)/256, 256, 0, stream>>>(target, attn_f, feats, gin);
  gemm_bf16_kernel<<<(Mq/512)*(D_/32), 256, LDSB_, stream>>>(
      gin, w1p, g_b1, hbuf, Mq, D_, KP_, 3, 0);
  gate_kernel<<<Mq, 256, 0, stream>>>(hbuf, g_w2, g_b2, gatev);

  // fuse + final layernorm; outputs: [B*T*D] out then [B*T] gate
  float* outp = (float*)d_out;
  final_kernel<<<Mq, 256, 0, stream>>>(target, attn_f, gatev, out_ln_w, out_ln_b,
                                       outp, outp + (size_t)Mq * D_);
}
